// GAT_3075196584582
// MI455X (gfx1250) — matrix operand order repaired and checked by reading, not compiled, not run on hardware
//
#include <hip/hip_runtime.h>

// ============================================================================
// CDNA5 (gfx1250, wave32) 4-layer GAT inference.
//
// Features are kept in bf16 in global memory (produced by the fused
// bias+ELU epilogue), so the WMMA GEMM loop is pure loads + v_wmma:
//   A frag = 2x global_load_b128 (contiguous k = 16h + s order),
//   B frag = 2x ds_load_b128 from a TDM-staged (tensor_load_to_lds) W tile.
//
// Per layer:
//   K0 gat_w2bf        : W f32 row-major -> bf16 column-major (once/layer)
//   K1 gat_gemm_wmma   : h = in @ W   (bf16 WMMA 16x16x32, f32 accum)
//   K2 gat_attcoef     : a_src/a_dst = <h, att>  per (node, head)
//   -- memset accum/m/denom --
//   K3 gat_edge_max    : segment-max of leakyrelu logits (ordered-uint max)
//   K4 gat_edge_exp    : alpha_e = exp(e - m[dst]); denom[dst] += alpha_e
//   K5 gat_scatter     : accum[dst] += h[src] * alpha/denom (float4+atomics)
//   K6 gat_bias_elu_bf : bias+ELU -> next layer's bf16 features
//      (layer 4: gat_bias_out -> f32 d_out)
//
// Roofline: edge phase dominates (~1.7GB/layer of gathers + f32 atomics) but
// all per-layer buffers are L2-resident (192MB), so this is L2-bound.
// ============================================================================

#define HIDC   64
#define NSLOPE 0.2f

typedef __bf16 bf16_t;
typedef __attribute__((ext_vector_type(16))) __bf16        v16bf;
typedef __attribute__((ext_vector_type(8)))  float         v8f;
typedef __attribute__((ext_vector_type(4)))  unsigned int  v4u;
typedef __attribute__((ext_vector_type(8)))  int           v8i;
typedef __attribute__((ext_vector_type(4)))  int           v4i;

__device__ __forceinline__ unsigned short f2bf_bits(float f) {
  union { float f; unsigned u; } x; x.f = f;
  unsigned r = x.u + 0x7FFFu + ((x.u >> 16) & 1u);   // round-to-nearest-even
  return (unsigned short)(r >> 16);
}

// monotone float <-> uint mapping so atomicMax(uint) == float max
__device__ __forceinline__ unsigned ford(float f) {
  union { float f; unsigned u; } x; x.f = f;
  return (x.u >> 31) ? ~x.u : (x.u | 0x80000000u);
}
__device__ __forceinline__ float funord(unsigned u) {
  union { float f; unsigned u; } x;
  x.u = (u & 0x80000000u) ? (u & 0x7FFFFFFFu) : ~u;
  return x.f;
}

// ---------------------------------------------------------------------------
// f32 -> bf16, 8 elements per thread (for the layer-1 input features)
// ---------------------------------------------------------------------------
__global__ void gat_f2bf8(const float* __restrict__ src,
                          uint4* __restrict__ dst, int n8)
{
  const int t = blockIdx.x * blockDim.x + threadIdx.x;
  if (t >= n8) return;
  const float4* s = (const float4*)(src + (size_t)t * 8);
  float4 a = s[0], b = s[1];
  uint4 o;
  o.x = (unsigned)f2bf_bits(a.x) | ((unsigned)f2bf_bits(a.y) << 16);
  o.y = (unsigned)f2bf_bits(a.z) | ((unsigned)f2bf_bits(a.w) << 16);
  o.z = (unsigned)f2bf_bits(b.x) | ((unsigned)f2bf_bits(b.y) << 16);
  o.w = (unsigned)f2bf_bits(b.z) | ((unsigned)f2bf_bits(b.w) << 16);
  dst[t] = o;
}

// ---------------------------------------------------------------------------
// K0: W [KIN,NC] f32 row-major  ->  wbf [NC][KIN] bf16 (column-major)
// ---------------------------------------------------------------------------
template<int KIN, int NC>
__global__ void gat_w2bf(const float* __restrict__ W,
                         unsigned short* __restrict__ wbf)
{
  const int i = blockIdx.x * 256 + threadIdx.x;
  if (i >= KIN * NC) return;
  const int nn = i / KIN, kk = i - nn * KIN;
  wbf[i] = f2bf_bits(W[(size_t)kk * NC + nn]);
}

// ---------------------------------------------------------------------------
// K1: tiled GEMM, one 16x64 C tile per wave, 8 waves/block.
//     W tile DMA'd into LDS by the Tensor Data Mover; A is bf16 in global.
// ---------------------------------------------------------------------------
template<int KIN, int NC>
__global__ void __launch_bounds__(256, 2)
gat_gemm_wmma(const unsigned short* __restrict__ xbf,
              const unsigned short* __restrict__ wbf,
              float* __restrict__ out, int n)
{
  extern __shared__ unsigned short ldsW[];            // [NC][KIN] bf16

  // ---- stage W via TDM: 2-D tensor, 2-byte elems, tile == tensor ----
  if (threadIdx.x < 32) {
    const unsigned lds_lo = (unsigned)(size_t)(&ldsW[0]); // flat LDS addr[31:0]
    const unsigned long long ga = (unsigned long long)(size_t)wbf;
    v4u g0;
    g0[0] = 1u;                                       // count=1 (valid D#)
    g0[1] = lds_lo;                                   // lds_addr
    g0[2] = (unsigned)(ga & 0xFFFFFFFFu);             // global_addr[31:0]
    g0[3] = (unsigned)((ga >> 32) & 0x01FFFFFFu) | (2u << 30); // [56:32]|type=2
    v8i g1;
    g1[0] = 0x00010000;                               // data_size=1 (2 bytes)
    g1[1] = (int)((KIN & 0xFFFF) << 16);              // tensor_dim0[15:0]
    g1[2] = (int)(((unsigned)KIN >> 16) | ((NC & 0xFFFF) << 16)); // td0hi|td1lo
    g1[3] = (int)(((unsigned)NC >> 16) | ((KIN & 0xFFFF) << 16)); // td1hi|tile0
    g1[4] = (int)(NC & 0xFFFF);                       // tile_dim1 (tile_dim2=0)
    g1[5] = KIN;                                      // tensor_dim0_stride lo
    g1[6] = 0;
    g1[7] = 0;
    v4i gz; gz[0] = gz[1] = gz[2] = gz[3] = 0;        // <=2D: groups 2/3 unused
#if __clang_major__ >= 23
    v8i gz8; for (int i = 0; i < 8; ++i) gz8[i] = 0;
    __builtin_amdgcn_tensor_load_to_lds(g0, g1, gz, gz, gz8, 0);
#else
    __builtin_amdgcn_tensor_load_to_lds(g0, g1, gz, gz, 0);
#endif
    __builtin_amdgcn_s_wait_tensorcnt(0);
  }
  __syncthreads();

  constexpr int colTiles = NC >> 6;                   // 4 (NC=256) or 1 (NC=64)
  constexpr int rowsPerBlock = (8 / colTiles) << 4;   // 32 or 128
  const int tid = threadIdx.x;
  const int wave = tid >> 5, lane = tid & 31;
  const int colTile = wave % colTiles;
  const int rowTile = wave / colTiles;
  const int mBase = blockIdx.x * rowsPerBlock + (rowTile << 4);
  if (mBase >= n) return;

  const int r0 = lane & 15;
  int arow = mBase + r0; if (arow >= n) arow = n - 1; // clamp tail reads
  const unsigned short* __restrict__ ap = xbf + (size_t)arow * KIN;
  const int aKoff = (lane < 16) ? 0 : 16;             // contiguous k order: k = 16h + s
  const int bKoff = (lane < 16) ? 0 : 16;             // contiguous k order: k = 16h + s (same as A)
  const int colBase = colTile << 6;

  v8f acc[4];
  #pragma unroll
  for (int t = 0; t < 4; ++t)
    #pragma unroll
    for (int i = 0; i < 8; ++i) acc[t][i] = 0.0f;

  union FV { uint4 u[2]; v16bf v; };

  #pragma unroll
  for (int k0 = 0; k0 < KIN; k0 += 32) {
    // A fragment: two aligned b128 loads (K groups [k0+aKoff..+7], [k0+aKoff+8..+15])
    FV av;
    av.u[0] = *(const uint4*)(ap + k0 + aKoff);
    av.u[1] = *(const uint4*)(ap + k0 + 8 + aKoff);
    // 4 B fragments from LDS, then the WMMA chain
    FV bb[4];
    #pragma unroll
    for (int t = 0; t < 4; ++t) {
      const uint4* bp =
          (const uint4*)&ldsW[(size_t)(colBase + (t << 4) + r0) * KIN + k0 + bKoff];
      bb[t].u[0] = bp[0]; bb[t].u[1] = bp[1];
    }
    #pragma unroll
    for (int t = 0; t < 4; ++t)
      acc[t] = __builtin_amdgcn_wmma_f32_16x16x32_bf16(
          false, av.v, false, bb[t].v, (short)0, acc[t], false, false);
  }

  // C/D layout: VGPR r -> M=r (lanes 0-15) / M=8+r (lanes 16-31)
  const int rowTop = mBase + ((lane >> 4) << 3);
  if (rowTop + 8 <= n) {                              // fast unguarded path
    #pragma unroll
    for (int t = 0; t < 4; ++t) {
      float* p = out + (size_t)rowTop * NC + colBase + (t << 4) + r0;
      #pragma unroll
      for (int rr = 0; rr < 8; ++rr) { *p = acc[t][rr]; p += NC; }
    }
  } else {
    #pragma unroll
    for (int t = 0; t < 4; ++t) {
      const int col = colBase + (t << 4) + r0;
      for (int rr = 0; rr < 8 && rowTop + rr < n; ++rr)
        out[(size_t)(rowTop + rr) * NC + col] = acc[t][rr];
    }
  }
}

// ---------------------------------------------------------------------------
// K2: per-(node,head) attention coefficients
// ---------------------------------------------------------------------------
template<int HEADS>
__global__ void gat_attcoef(const float* __restrict__ h,
                            const float* __restrict__ att_s,
                            const float* __restrict__ att_d,
                            float* __restrict__ asrc, float* __restrict__ adst,
                            int n)
{
  const int t = blockIdx.x * blockDim.x + threadIdx.x;
  if (t >= n * HEADS) return;
  const int nd = t / HEADS, hh = t - nd * HEADS;
  const float* hp = h + (size_t)nd * (HEADS * HIDC) + hh * HIDC;
  const float* sp = att_s + hh * HIDC;
  const float* dp = att_d + hh * HIDC;
  float as = 0.f, ad = 0.f;
  #pragma unroll
  for (int c = 0; c < HIDC; c += 4) {
    float4 hv = *(const float4*)(hp + c);
    float4 sv = *(const float4*)(sp + c);
    float4 dv = *(const float4*)(dp + c);
    as += hv.x * sv.x + hv.y * sv.y + hv.z * sv.z + hv.w * sv.w;
    ad += hv.x * dv.x + hv.y * dv.y + hv.z * dv.z + hv.w * dv.w;
  }
  asrc[t] = as; adst[t] = ad;
}

// ---------------------------------------------------------------------------
// K3: segment max of leaky-relu logits (incl. self loops)
// ---------------------------------------------------------------------------
template<int HEADS>
__global__ void gat_edge_max(const long long* __restrict__ ei,
                             const float* __restrict__ asrc,
                             const float* __restrict__ adst,
                             unsigned* __restrict__ mmax, int E, int n)
{
  const int e = blockIdx.x * blockDim.x + threadIdx.x;
  if (e >= E + n) return;
  int s, d;
  if (e < E) { s = (int)ei[e]; d = (int)ei[(size_t)E + e]; }
  else       { s = d = e - E; }
  #pragma unroll
  for (int hh = 0; hh < HEADS; ++hh) {
    float v = asrc[s * HEADS + hh] + adst[d * HEADS + hh];
    v = v > 0.f ? v : NSLOPE * v;
    atomicMax(&mmax[d * HEADS + hh], ford(v));
  }
}

// ---------------------------------------------------------------------------
// K4: alpha_e = exp(e - m[dst]);  denom[dst] += alpha_e
// ---------------------------------------------------------------------------
template<int HEADS>
__global__ void gat_edge_exp(const long long* __restrict__ ei,
                             const float* __restrict__ asrc,
                             const float* __restrict__ adst,
                             const unsigned* __restrict__ mmax,
                             float* __restrict__ alpha,
                             float* __restrict__ denom, int E, int n)
{
  const int e = blockIdx.x * blockDim.x + threadIdx.x;
  if (e >= E + n) return;
  int s, d;
  if (e < E) { s = (int)ei[e]; d = (int)ei[(size_t)E + e]; }
  else       { s = d = e - E; }
  #pragma unroll
  for (int hh = 0; hh < HEADS; ++hh) {
    float v = asrc[s * HEADS + hh] + adst[d * HEADS + hh];
    v = v > 0.f ? v : NSLOPE * v;
    float ex = __expf(v - funord(mmax[d * HEADS + hh]));
    alpha[(size_t)e * HEADS + hh] = ex;
    atomicAdd(&denom[d * HEADS + hh], ex);
  }
}

// ---------------------------------------------------------------------------
// K5: accum[dst] += h[src] * alpha/denom   -- one float4 chunk per thread
// ---------------------------------------------------------------------------
template<int HEADS, int NC>
__global__ void gat_scatter(const long long* __restrict__ ei,
                            const float* __restrict__ h,
                            const float* __restrict__ alpha,
                            const float* __restrict__ denom,
                            float* __restrict__ out, int E, int n)
{
  constexpr int CHUNKS = NC >> 2;                     // power of two
  const long long gid = (long long)blockIdx.x * blockDim.x + threadIdx.x;
  if (gid >= (long long)(E + n) * CHUNKS) return;
  const int c4 = (int)(gid & (CHUNKS - 1));
  const int e  = (int)(gid >> (NC == 256 ? 6 : 4));
  int s, d;
  if (e < E) { s = (int)ei[e]; d = (int)ei[(size_t)E + e]; }
  else       { s = d = e - E; }
  const int hh = (c4 << 2) / HIDC;                    // constant-folded
  const float al = alpha[(size_t)e * HEADS + hh] / denom[d * HEADS + hh];
  const float4 hv = *(const float4*)(h + (size_t)s * NC + (c4 << 2));
  float* op = out + (size_t)d * NC + (c4 << 2);
  atomicAdd(op + 0, hv.x * al);
  atomicAdd(op + 1, hv.y * al);
  atomicAdd(op + 2, hv.z * al);
  atomicAdd(op + 3, hv.w * al);
}

// ---------------------------------------------------------------------------
// K6: bias + ELU, fused f32 -> bf16 (two elements / thread, packed store)
// ---------------------------------------------------------------------------
template<int NC>
__global__ void gat_bias_elu_bf(const float* __restrict__ acc,
                                const float* __restrict__ bias,
                                unsigned* __restrict__ featOut, int n)
{
  const int t = blockIdx.x * blockDim.x + threadIdx.x;   // bf16-pair index
  if (t >= (n * NC) >> 1) return;
  const int t2 = t << 1;
  float v0 = acc[t2]     + bias[t2 & (NC - 1)];
  float v1 = acc[t2 + 1] + bias[(t2 + 1) & (NC - 1)];
  v0 = v0 > 0.f ? v0 : __expf(v0) - 1.0f;
  v1 = v1 > 0.f ? v1 : __expf(v1) - 1.0f;
  featOut[t] = (unsigned)f2bf_bits(v0) | ((unsigned)f2bf_bits(v1) << 16);
}

template<int NC>
__global__ void gat_bias_out(const float* __restrict__ in,
                             const float* __restrict__ bias,
                             float* __restrict__ out, int n)
{
  const int t = blockIdx.x * blockDim.x + threadIdx.x;
  if (t >= n * NC) return;
  out[t] = in[t] + bias[t & (NC - 1)];
}

// ---------------------------------------------------------------------------
// Host orchestration
// ---------------------------------------------------------------------------
template<int HEADS, int NC>
static void run_layer(const unsigned short* featIn, const long long* ei,
                      const float* W, const float* att_s, const float* att_d,
                      const float* bias,
                      float* hbuf, float* accum, unsigned short* wbf,
                      float* asrc, float* adst, unsigned* mmax,
                      float* denom, float* alpha,
                      int N, int E,
                      unsigned short* featOut, float* finalOut,
                      hipStream_t stream)
{
  constexpr int KIN = 256;
  gat_w2bf<KIN, NC><<<(KIN * NC + 255) / 256, 256, 0, stream>>>(W, wbf);

  constexpr int rowsPerBlock = (8 / (NC >> 6)) << 4;
  const unsigned gGemm = (unsigned)((N + rowsPerBlock - 1) / rowsPerBlock);
  const size_t shW = (size_t)KIN * NC * sizeof(unsigned short);
  gat_gemm_wmma<KIN, NC><<<gGemm, 256, shW, stream>>>(featIn, wbf, hbuf, N);

  const int nh = N * HEADS;
  gat_attcoef<HEADS><<<(unsigned)((nh + 255) / 256), 256, 0, stream>>>(
      hbuf, att_s, att_d, asrc, adst, N);

  hipMemsetAsync(accum, 0, (size_t)N * NC * sizeof(float), stream);
  hipMemsetAsync(mmax, 0, (size_t)nh * sizeof(unsigned), stream);
  hipMemsetAsync(denom, 0, (size_t)nh * sizeof(float), stream);

  const int tot = E + N;
  gat_edge_max<HEADS><<<(unsigned)((tot + 255) / 256), 256, 0, stream>>>(
      ei, asrc, adst, mmax, E, N);
  gat_edge_exp<HEADS><<<(unsigned)((tot + 255) / 256), 256, 0, stream>>>(
      ei, asrc, adst, mmax, alpha, denom, E, N);

  const long long stot = (long long)tot * (NC / 4);
  gat_scatter<HEADS, NC><<<(unsigned)((stot + 255) / 256), 256, 0, stream>>>(
      ei, hbuf, alpha, denom, accum, E, N);

  if (finalOut) {
    const int ne = N * NC;
    gat_bias_out<NC><<<(unsigned)((ne + 255) / 256), 256, 0, stream>>>(
        accum, bias, finalOut, N);
  } else {
    const int np = (N * NC) / 2;
    gat_bias_elu_bf<NC><<<(unsigned)((np + 255) / 256), 256, 0, stream>>>(
        accum, bias, (unsigned*)featOut, N);
  }
}

extern "C" void kernel_launch(void* const* d_in, const int* in_sizes, int n_in,
                              void* d_out, int out_size, void* d_ws, size_t ws_size,
                              hipStream_t stream)
{
  (void)n_in; (void)out_size; (void)ws_size;
  const float*     x  = (const float*)d_in[0];
  const long long* ei = (const long long*)d_in[1];    // int64 edge_index [2,E]
  const float* W1  = (const float*)d_in[2];
  const float* as1 = (const float*)d_in[3];
  const float* ad1 = (const float*)d_in[4];
  const float* b1  = (const float*)d_in[5];
  const float* W2  = (const float*)d_in[6];
  const float* as2 = (const float*)d_in[7];
  const float* ad2 = (const float*)d_in[8];
  const float* b2  = (const float*)d_in[9];
  const float* W3  = (const float*)d_in[10];
  const float* as3 = (const float*)d_in[11];
  const float* ad3 = (const float*)d_in[12];
  const float* b3  = (const float*)d_in[13];
  const float* W4  = (const float*)d_in[14];
  const float* as4 = (const float*)d_in[15];
  const float* ad4 = (const float*)d_in[16];
  const float* b4  = (const float*)d_in[17];

  const int N = in_sizes[0] / 256;                    // F_IN = 256
  const int E = in_sizes[1] / 2;

  // workspace: H | accum | asrc | adst | mmax | denom | alpha | featA | featB | wbf
  float* ws = (float*)d_ws;
  const size_t fN256 = (size_t)N * 256;
  float*    bufH  = ws;
  float*    accum = bufH + fN256;
  float*    asrc  = accum + fN256;
  float*    adst  = asrc + (size_t)N * 4;
  unsigned* mmax  = (unsigned*)(adst + (size_t)N * 4);
  float*    denom = (float*)mmax + (size_t)N * 4;
  float*    alpha = denom + (size_t)N * 4;
  unsigned short* featA = (unsigned short*)(alpha + (size_t)(E + N) * 4);
  unsigned short* featB = featA + fN256;
  unsigned short* wbf   = featB + fN256;

  // x (f32) -> featA (bf16)
  const int n8 = (int)(fN256 / 8);
  gat_f2bf8<<<(unsigned)((n8 + 255) / 256), 256, 0, stream>>>(
      x, (uint4*)featA, n8);

  // layer 1: featA -> featB   (4 heads x 64, concat, ELU)
  run_layer<4, 256>(featA, ei, W1, as1, ad1, b1, bufH, accum, wbf,
                    asrc, adst, mmax, denom, alpha, N, E, featB, nullptr, stream);
  // layer 2: featB -> featA
  run_layer<4, 256>(featB, ei, W2, as2, ad2, b2, bufH, accum, wbf,
                    asrc, adst, mmax, denom, alpha, N, E, featA, nullptr, stream);
  // layer 3: featA -> featB
  run_layer<4, 256>(featA, ei, W3, as3, ad3, b3, bufH, accum, wbf,
                    asrc, adst, mmax, denom, alpha, N, E, featB, nullptr, stream);
  // layer 4: featB -> d_out[N,64]  (1 head, mean==identity, bias, no ELU)
  run_layer<1, 64>(featB, ei, W4, as4, ad4, b4, bufH, accum, wbf,
                   asrc, adst, mmax, denom, alpha, N, E,
                   nullptr, (float*)d_out, stream);
}
